// EmissionModel_22677427323535
// MI455X (gfx1250) — compile-verified, matrix operand order checked by reading, not run on hardware
//
#include <hip/hip_runtime.h>
#include <hip/hip_bf16.h>
#include <cstddef>

typedef _Float16 v16h __attribute__((ext_vector_type(16)));
typedef _Float16 v8h  __attribute__((ext_vector_type(8)));
typedef float    v8f  __attribute__((ext_vector_type(8)));

#define NV 14
#define ED 32
#define NVIEW 3
#define KSTATES 16384
#define NB 32

// ---------------- B-fragment index decode (shared by prep + compute) -------
// B element (k,n) of a 32x16 (KxN) f16 tile:
//   g = (k%32)/8 ; lane = (n%16) + 16*(g%2) ; half-idx = (g/2)*8 + k%8
// Storage: frag[((kt*NT + nt)*32 + lane)*16 + idx]
template<int KT, int NT>
__device__ __forceinline__ void bdec(int gid, int& v, int& k, int& n) {
  int idx = gid & 15, lane = (gid >> 4) & 31, fn = gid >> 9;
  int nt = fn % NT; int t = fn / NT; int kt = t % KT; v = t / KT;
  int p = lane >> 4, nn = lane & 15, run = idx >> 3, kk = idx & 7;
  k = kt * 32 + (run * 2 + p) * 8 + kk;
  n = nt * 16 + nn;
}

// ---------------- prep kernels: swizzle weights into fragment order --------
__global__ void prep_emb(const float* src, _Float16* dst) {
  int i = blockIdx.x * blockDim.x + threadIdx.x;
  if (i < NV * 2 * ED) dst[i] = (_Float16)src[i];
}

__global__ void prep_linW(const float* lin_w, _Float16* dst) {  // (3,448,64)
  int gid = blockIdx.x * blockDim.x + threadIdx.x;
  if (gid >= 3 * 14 * 4 * 512) return;
  int v, k, n; bdec<14, 4>(gid, v, k, n);
  dst[gid] = (_Float16)lin_w[((size_t)v * 448 + k) * 64 + n];
}

__global__ void prep_W1(const float* w1, _Float16* dst) {  // (3,64,32,3,3)->(3,64,288)
  int gid = blockIdx.x * blockDim.x + threadIdx.x;
  if (gid >= 3 * 2 * 18 * 512) return;
  int v, k, n; bdec<2, 18>(gid, v, k, n);
  dst[gid] = (_Float16)w1[((size_t)v * 64 + k) * 288 + n];
}

__global__ void prep_W2(const float* w2, _Float16* dst) {  // (3,32,16,3,3)
  int gid = blockIdx.x * blockDim.x + threadIdx.x;
  if (gid >= 3 * 9 * 25 * 512) return;
  int v, k, n; bdec<9, 25>(gid, v, k, n);
  int ci = k / 9, r9 = k % 9, pp = r9 / 3, qq = r9 % 3;
  int co = n / 25, r25 = n % 25, y = r25 / 5, xq = r25 % 5;
  int dy = y - pp, dx = xq - qq;
  float val = 0.f;
  if (dy >= 0 && dy < 3 && dx >= 0 && dx < 3)
    val = w2[(((size_t)(v * 32 + ci) * 16 + co) * 3 + dy) * 3 + dx];
  dst[gid] = (_Float16)val;
}

__global__ void prep_W3(const float* w3, _Float16* dst) {  // (3,16,1,3,3)
  int gid = blockIdx.x * blockDim.x + threadIdx.x;
  if (gid >= 3 * 13 * 4 * 512) return;
  int v, k, n; bdec<13, 4>(gid, v, k, n);
  float val = 0.f;
  if (k < 400 && n < 49) {
    int ci = k / 25, r = k % 25, pp = r / 5, qq = r % 5;
    int y = n / 7, xq = n % 7;
    int dy = y - pp, dx = xq - qq;
    if (dy >= 0 && dy < 3 && dx >= 0 && dx < 3)
      val = w3[(size_t)(v * 16 + ci) * 9 + dy * 3 + dx];
  }
  dst[gid] = (_Float16)val;
}

__global__ void prep_X(const float* x, _Float16* dst) {  // x:(32,3,7,7); B=X^T (49pad64 x 32)
  int gid = blockIdx.x * blockDim.x + threadIdx.x;
  if (gid >= 3 * 2 * 2 * 512) return;
  int v, k, n; bdec<2, 2>(gid, v, k, n);
  float val = (k < 49) ? x[((size_t)n * 3 + v) * 49 + k] : 0.f;
  dst[gid] = (_Float16)val;
}

__global__ void prep_xx(const float* x, float* xx) {
  int gid = blockIdx.x * blockDim.x + threadIdx.x;
  if (gid >= 3 * 32) return;
  int v = gid / 32, b = gid % 32;
  float s = 0.f;
  for (int h = 0; h < 49; ++h) {
    float t = x[((size_t)b * 3 + v) * 49 + h];
    s += t * t;
  }
  xx[gid] = s;
}

// ---------------- main compute kernel --------------------------------------
__device__ __forceinline__ v16h ldsA(const _Float16* buf, int stride, int kt, int lane) {
  const int m = lane & 15, p = lane >> 4;
  const _Float16* r0 = buf + m * stride + kt * 32 + p * 8;
  v8h lo = *(const v8h*)r0;
  v8h hi = *(const v8h*)(r0 + 16);
  v16h a;
#pragma unroll
  for (int i = 0; i < 8; ++i) { a[i] = lo[i]; a[i + 8] = hi[i]; }
  return a;
}

__device__ __forceinline__ void gemm_stage(
    const _Float16* src, int sStride, _Float16* dst, int dStride,
    const _Float16* Bfrag, int KT, int NT,
    const float* bias, int biasDiv, bool elu, int lane, float* dstF32) {
  const int p = lane >> 4, nn = lane & 15;
  for (int nt = 0; nt < NT; ++nt) {
    const int col = nt * 16 + nn;
    const float bv = bias[col / biasDiv];
    v8f c;
#pragma unroll
    for (int r = 0; r < 8; ++r) c[r] = bv;
    for (int kt = 0; kt < KT; ++kt) {
      v16h a = ldsA(src, sStride, kt, lane);
      v16h b = *(const v16h*)(Bfrag + ((size_t)(kt * NT + nt) * 32 + lane) * 16);
      c = __builtin_amdgcn_wmma_f32_16x16x32_f16(false, a, false, b, (short)0, c, false, false);
    }
#pragma unroll
    for (int r = 0; r < 8; ++r) {
      float d = c[r];
      if (elu) d = d > 0.f ? d : (__expf(d) - 1.f);
      const int m = r + 8 * p;
      dst[m * dStride + col] = (_Float16)d;
      if (dstF32) dstF32[m * 64 + col] = d;
    }
  }
}

__global__ __launch_bounds__(64) void emission_main(
    const float* lin_b, const float* b1, const float* b2, const float* b3,
    const _Float16* emb16, const _Float16* linWfrag, const _Float16* W1frag,
    const _Float16* W2frag, const _Float16* W3frag, const _Float16* Xfrag,
    const float* xx, float* out) {
  __shared__ __align__(32) _Float16 bufA[2][16 * 416];  // wide activations (A2)
  __shared__ __align__(32) _Float16 bufB[2][16 * 288];  // A1 / M(f16)
  __shared__ __align__(16) float    bufC[2][16 * 64];   // M(f32) for sum-of-squares
  __shared__ float ssBuf[2][16];

  const int lane = threadIdx.x & 31;
  const int w    = threadIdx.x >> 5;
  const int tile = blockIdx.x * 2 + w;
  const int s0   = tile * 16;
  _Float16* A  = bufA[w];
  _Float16* Bb = bufB[w];
  float*    Cc = bufC[w];

  // zero the K-padding columns 400..415 of bufA once (never written by stages)
  for (int i = lane; i < 16 * 16; i += 32) {
    int m = i >> 4, c = i & 15;
    A[m * 416 + 400 + c] = (_Float16)0.f;
  }
  __syncthreads();

  for (int v = 0; v < NVIEW; ++v) {
    // ---- stage 1: H (16x64) = Z(16x448 via bit-gathered embeddings) @ linW + lin_b
    {
      const int m = lane & 15, p = lane >> 4, nn = lane & 15;
      const int state = s0 + m;
      for (int nt = 0; nt < 4; ++nt) {
        const int col = nt * 16 + nn;
        const float bv = lin_b[v * 64 + col];
        v8f c;
#pragma unroll
        for (int r = 0; r < 8; ++r) c[r] = bv;
        for (int kt = 0; kt < 14; ++kt) {
          const int bit = (state >> (13 - kt)) & 1;  // idcs: variable 0 = MSB
          const _Float16* e = emb16 + (kt * 2 + bit) * 32 + p * 8;
          v8h lo = *(const v8h*)e;
          v8h hi = *(const v8h*)(e + 16);
          v16h a;
#pragma unroll
          for (int i = 0; i < 8; ++i) { a[i] = lo[i]; a[i + 8] = hi[i]; }
          v16h b = *(const v16h*)(linWfrag + ((size_t)((v * 14 + kt) * 4 + nt) * 32 + lane) * 16);
          c = __builtin_amdgcn_wmma_f32_16x16x32_f16(false, a, false, b, (short)0, c, false, false);
        }
#pragma unroll
        for (int r = 0; r < 8; ++r) A[(r + 8 * p) * 416 + col] = (_Float16)c[r];
      }
    }
    __syncthreads();
    // ---- stage 2: A1 (16x288) = ELU(H @ W1 + b1) ; convT 1x1 -> 3x3
    gemm_stage(A, 416, Bb, 288, W1frag + (size_t)v * 2 * 18 * 512, 2, 18,
               b1 + v * 32, 9, true, lane, nullptr);
    __syncthreads();
    // ---- stage 3: A2 (16x400, pad 416) = ELU(A1 @ W2eff + b2) ; 3x3 -> 5x5
    gemm_stage(Bb, 288, A, 416, W2frag + (size_t)v * 9 * 25 * 512, 9, 25,
               b2 + v * 16, 25, true, lane, nullptr);
    __syncthreads();
    // ---- stage 4: M (16x49, pad 64) = A2 @ W3eff + b3 ; 5x5 -> 7x7
    gemm_stage(A, 416, Bb, 288, W3frag + (size_t)v * 13 * 4 * 512, 13, 4,
               b3 + v, 64, false, lane, Cc);
    __syncthreads();
    // ---- sum of squares per state (f32)
    if (lane < 16) {
      float s = 0.f;
#pragma unroll
      for (int h = 0; h < 49; ++h) { float m = Cc[lane * 64 + h]; s += m * m; }
      ssBuf[w][lane] = s;
    }
    __syncthreads();
    // ---- stage 5: dot (16x32) = M @ X^T ; out = dot - 0.5*(ss + xx)
    {
      const int p = lane >> 4, nn = lane & 15;
      for (int nt = 0; nt < 2; ++nt) {
        v8f c;
#pragma unroll
        for (int r = 0; r < 8; ++r) c[r] = 0.f;
        for (int kt = 0; kt < 2; ++kt) {
          v16h a = ldsA(Bb, 288, kt, lane);
          v16h b = *(const v16h*)(Xfrag + ((size_t)((v * 2 + kt) * 2 + nt) * 32 + lane) * 16);
          c = __builtin_amdgcn_wmma_f32_16x16x32_f16(false, a, false, b, (short)0, c, false, false);
        }
        const int bcol = nt * 16 + nn;
        const float xv = xx[v * 32 + bcol];
#pragma unroll
        for (int r = 0; r < 8; ++r) {
          const int s = r + 8 * p;
          out[((size_t)bcol * KSTATES + (s0 + s)) * 3 + v] = c[r] - 0.5f * (ssBuf[w][s] + xv);
        }
      }
    }
    __syncthreads();
  }
}

// ---------------- host launcher --------------------------------------------
extern "C" void kernel_launch(void* const* d_in, const int* in_sizes, int n_in,
                              void* d_out, int out_size, void* d_ws, size_t ws_size,
                              hipStream_t stream) {
  const float* x     = (const float*)d_in[0];   // (32,3,7,7)
  const float* lemb  = (const float*)d_in[1];   // (14,2,32)
  const float* lin_w = (const float*)d_in[2];   // (3,448,64)
  const float* lin_b = (const float*)d_in[3];   // (3,64)
  const float* w1    = (const float*)d_in[4];   // (3,64,32,3,3)
  const float* b1    = (const float*)d_in[5];   // (3,32)
  const float* w2    = (const float*)d_in[6];   // (3,32,16,3,3)
  const float* b2    = (const float*)d_in[7];   // (3,16)
  const float* w3    = (const float*)d_in[8];   // (3,16,1,3,3)
  const float* b3    = (const float*)d_in[9];   // (3,1)
  float* out = (float*)d_out;

  // workspace carve (all fragment arrays 32B-aligned by construction)
  _Float16* emb16    = (_Float16*)d_ws;             // 896 halves
  _Float16* linWfrag = emb16 + 896;                 // 3*14*4*512 = 86016
  _Float16* W1frag   = linWfrag + 86016;            // 3*2*18*512 = 55296
  _Float16* W2frag   = W1frag + 55296;              // 3*9*25*512 = 345600
  _Float16* W3frag   = W2frag + 345600;             // 3*13*4*512 = 79872
  _Float16* Xfrag    = W3frag + 79872;              // 3*2*2*512  = 6144
  float*    xx       = (float*)(Xfrag + 6144);      // 96 floats

  prep_emb <<<dim3(4),    dim3(256), 0, stream>>>(lemb,  emb16);
  prep_linW<<<dim3(336),  dim3(256), 0, stream>>>(lin_w, linWfrag);
  prep_W1  <<<dim3(216),  dim3(256), 0, stream>>>(w1,    W1frag);
  prep_W2  <<<dim3(1350), dim3(256), 0, stream>>>(w2,    W2frag);
  prep_W3  <<<dim3(312),  dim3(256), 0, stream>>>(w3,    W3frag);
  prep_X   <<<dim3(24),   dim3(256), 0, stream>>>(x,     Xfrag);
  prep_xx  <<<dim3(1),    dim3(256), 0, stream>>>(x,     xx);

  emission_main<<<dim3(512), dim3(64), 0, stream>>>(
      lin_b, b1, b2, b3, emb16, linWfrag, W1frag, W2frag, W3frag, Xfrag, xx, out);
}
